// SPRTFlipLinear_15058155339842
// MI455X (gfx1250) — compile-verified, matrix operand order checked by reading, not run on hardware
//
#include <hip/hip_runtime.h>
#include <stdint.h>

// ---------------------------------------------------------------------------
// Ternary-dequant GEMM for MI455X (gfx1250, wave32, WMMA, async-to-LDS).
//   y[8192,4096] = x[8192,4096] * W^T,  W[n,k] = ternary[n,k]*scales[n*32+k/128]
// Compute-bound at f16 WMMA (~275 GFLOP vs ~12us HBM). Element-wise work
// (f32->f16 cvt, int8 dequant) is hoisted into memory-bound pre-passes into
// d_ws; the main kernel is a double-buffered pipeline:
//   global_load_async_to_lds_b128 (ASYNCcnt) -> barrier -> ds_load_b128 ->
//   16x v_wmma_f32_16x16x32_f16 per wave per k-stage (64x64 wave tile).
// ---------------------------------------------------------------------------

typedef __attribute__((ext_vector_type(16))) _Float16 v16h;
typedef __attribute__((ext_vector_type(8)))  _Float16 v8h;
typedef __attribute__((ext_vector_type(8)))  float    v8f;
typedef __attribute__((ext_vector_type(4)))  float    v4f;
typedef __attribute__((ext_vector_type(16))) int8_t   v16b;

#define M_DIM 8192
#define N_DIM 4096
#define K_DIM 4096
#define GROUP 128
#define BM 256        // block tile rows   (4 waves x 64)
#define BN 128        // block tile cols   (2 waves x 64)
#define BK 32
#define LDS_PAD 40    // 32 data halves + 8 pad = 80B rows: 16B-aligned, 20-bank stride
#define STAGES (K_DIM / BK)

__device__ __forceinline__ void async_wait0() {
#if __has_builtin(__builtin_amdgcn_s_wait_asynccnt)
  __builtin_amdgcn_s_wait_asynccnt(0);
#else
  asm volatile("s_wait_asynccnt 0x0" ::: "memory");
#endif
}

// GLOBAL_LOAD_ASYNC_TO_LDS_B128: vdst VGPR = LDS byte address, vaddr = 64-bit
// global address ("GV" mode). Tracked by ASYNCcnt.
__device__ __forceinline__ void async_ld_b128(uint32_t lds_addr, const void* gaddr) {
  asm volatile("global_load_async_to_lds_b128 %0, %1, off"
               :: "v"(lds_addr), "v"((uint64_t)(uintptr_t)gaddr)
               : "memory");
}

// ---------------------------------------------------------------------------
// Pre-pass 1: x f32 -> f16 (memory-bound)
// ---------------------------------------------------------------------------
__global__ __launch_bounds__(256)
void cvt_x_f16(const float* __restrict__ x, _Float16* __restrict__ xw) {
  const size_t i = ((size_t)blockIdx.x * 256 + threadIdx.x) * 8;
  const v4f a = *reinterpret_cast<const v4f*>(x + i);
  const v4f b = *reinterpret_cast<const v4f*>(x + i + 4);
  v8h h;
#pragma unroll
  for (int j = 0; j < 4; ++j) { h[j] = (_Float16)a[j]; h[j + 4] = (_Float16)b[j]; }
  *reinterpret_cast<v8h*>(xw + i) = h;
}

// ---------------------------------------------------------------------------
// Pre-pass 2: W = ternary * group scale -> f16 (exact: 16-chunks never cross
// a 128-wide group boundary)
// ---------------------------------------------------------------------------
__global__ __launch_bounds__(256)
void dequant_w_f16(const int8_t* __restrict__ tern, const float* __restrict__ scales,
                   _Float16* __restrict__ ww) {
  const size_t i = ((size_t)blockIdx.x * 256 + threadIdx.x) * 16;
  const v16b w = *reinterpret_cast<const v16b*>(tern + i);
  const size_t n = i >> 12;                 // row (out feature), K=4096
  const int    k = (int)(i & (K_DIM - 1));
  const float  s = scales[n * (K_DIM / GROUP) + (k >> 7)];
  v8h h0, h1;
#pragma unroll
  for (int j = 0; j < 8; ++j) {
    h0[j] = (_Float16)(s * (float)w[j]);
    h1[j] = (_Float16)(s * (float)w[j + 8]);
  }
  *reinterpret_cast<v8h*>(ww + i)     = h0;
  *reinterpret_cast<v8h*>(ww + i + 8) = h1;
}

// ---------------------------------------------------------------------------
// Main GEMM: 256x128 block tile, 8 waves (4x2), 64x64 per wave ->
// 16 WMMA per 16 ds_load_b128 per stage. Double-buffered async-to-LDS.
// ---------------------------------------------------------------------------
__global__ __launch_bounds__(256)
void gemm_f16_async(const _Float16* __restrict__ xw,
                    const _Float16* __restrict__ ww,
                    float* __restrict__ out) {
  __shared__ alignas(16) _Float16 sA[2][BM * LDS_PAD];  // 40 KB
  __shared__ alignas(16) _Float16 sB[2][BN * LDS_PAD];  // 20 KB

  const int tid  = threadIdx.x;
  const int lane = tid & 31;
  const int wave = tid >> 5;
  const int wm   = wave >> 1;           // 0..3 : 64-row strip of 256
  const int wn   = wave & 1;            // 0..1 : 64-col strip of 128
  const size_t m0 = (size_t)blockIdx.y * BM;
  const size_t n0 = (size_t)blockIdx.x * BN;

  // Generic shared pointers carry the LDS byte address in their low 32 bits.
  const uint32_t ldsA = (uint32_t)(uintptr_t)&sA[0][0];
  const uint32_t ldsB = (uint32_t)(uintptr_t)&sB[0][0];

  // copy coords: chunk = 8 halves (16B); 4 chunks per 32-half row.
  // A: 256 rows * 4 = 1024 chunks -> 4/thread; B: 128 * 4 = 512 -> 2/thread.
  const int lrow = tid >> 2;            // 0..63
  const int lcol = (tid & 3) << 3;      // 0,8,16,24

  auto issue_stage = [&](int kt, int buf) {
#pragma unroll
    for (int it = 0; it < 4; ++it) {
      const int row = it * 64 + lrow;
      async_ld_b128(ldsA + (uint32_t)(2 * (buf * BM * LDS_PAD + row * LDS_PAD + lcol)),
                    xw + (m0 + row) * K_DIM + kt + lcol);
    }
#pragma unroll
    for (int it = 0; it < 2; ++it) {
      const int row = it * 64 + lrow;
      async_ld_b128(ldsB + (uint32_t)(2 * (buf * BN * LDS_PAD + row * LDS_PAD + lcol)),
                    ww + (n0 + row) * K_DIM + kt + lcol);
    }
  };

  // fragment LDS coords (ISA 7.12.2 16-bit A/B 16x32 layout):
  // lane L holds row/col L%16, K chunks {fk..fk+7} and {fk+16..fk+23}
  const int frow = lane & 15;
  const int fk   = (lane >> 4) << 3;    // 0 or 8

  v8f acc[4][4] = {};                   // 128 accumulator VGPRs / lane

  issue_stage(0, 0);

  for (int s = 0; s < STAGES; ++s) {
    async_wait0();        // my stage-s async copies are in LDS
    __syncthreads();      // everyone's are; buf[(s+1)&1] readers (stage s-1) done

    if (s + 1 < STAGES) issue_stage((s + 1) * BK, (s + 1) & 1);

    const _Float16* bufA = &sA[s & 1][0];
    const _Float16* bufB = &sB[s & 1][0];

    v16h aF[4];
#pragma unroll
    for (int mi = 0; mi < 4; ++mi) {
      const _Float16* p = bufA + (wm * 64 + mi * 16 + frow) * LDS_PAD + fk;
      v8h lo = *reinterpret_cast<const v8h*>(p);
      v8h hi = *reinterpret_cast<const v8h*>(p + 16);
      aF[mi] = __builtin_shufflevector(lo, hi, 0,1,2,3,4,5,6,7,8,9,10,11,12,13,14,15);
    }
    v16h bF[4];
#pragma unroll
    for (int ni = 0; ni < 4; ++ni) {
      const _Float16* p = bufB + (wn * 64 + ni * 16 + frow) * LDS_PAD + fk;
      v8h lo = *reinterpret_cast<const v8h*>(p);
      v8h hi = *reinterpret_cast<const v8h*>(p + 16);
      bF[ni] = __builtin_shufflevector(lo, hi, 0,1,2,3,4,5,6,7,8,9,10,11,12,13,14,15);
    }

#pragma unroll
    for (int mi = 0; mi < 4; ++mi)
#pragma unroll
      for (int ni = 0; ni < 4; ++ni)
        acc[mi][ni] = __builtin_amdgcn_wmma_f32_16x16x32_f16(
            false, aF[mi], false, bF[ni], (short)0, acc[mi][ni], false, false);
  }

  // writeback: C layout = VGPR r -> (M = r + 8*(lane/16), N = lane%16)
#pragma unroll
  for (int mi = 0; mi < 4; ++mi) {
#pragma unroll
    for (int ni = 0; ni < 4; ++ni) {
      const size_t row_b = m0 + wm * 64 + mi * 16 + ((lane >> 4) << 3);
      const size_t col   = n0 + wn * 64 + ni * 16 + (lane & 15);
#pragma unroll
      for (int r = 0; r < 8; ++r)
        out[(row_b + r) * N_DIM + col] = acc[mi][ni][r];
    }
  }
}

// ---------------------------------------------------------------------------
// Fallback (no workspace): in-loop dequant, 128x128 tile, 8 waves (4x2),
// 32x64 per wave.
// ---------------------------------------------------------------------------
__global__ __launch_bounds__(256)
void ternary_linear_wmma(const float*  __restrict__ x,
                         const int8_t* __restrict__ tern,
                         const float*  __restrict__ scales,
                         float*        __restrict__ out) {
  __shared__ alignas(16) _Float16 sA[128 * LDS_PAD];
  __shared__ alignas(16) _Float16 sB[128 * LDS_PAD];

  const int tid  = threadIdx.x;
  const int lane = tid & 31;
  const int wave = tid >> 5;
  const int wm   = wave >> 1;
  const int wn   = wave & 1;
  const size_t m0 = (size_t)blockIdx.y * 128;
  const size_t n0 = (size_t)blockIdx.x * 128;

  v8f acc[2][4] = {};
  const int bN    = tid >> 1;
  const int bHalf = (tid & 1) << 4;
  const int frow = lane & 15;
  const int fk   = (lane >> 4) << 3;

  for (int kt = 0; kt < K_DIM; kt += BK) {
#pragma unroll
    for (int it = 0; it < 4; ++it) {
      const int e   = it * 1024 + tid * 4;
      const int row = e >> 5;
      const int col = e & 31;
      const v4f xv = *reinterpret_cast<const v4f*>(x + (m0 + row) * K_DIM + kt + col);
      _Float16* dst = &sA[row * LDS_PAD + col];
#pragma unroll
      for (int j = 0; j < 4; ++j) dst[j] = (_Float16)xv[j];
    }
    {
      const v16b w = *reinterpret_cast<const v16b*>(tern + (n0 + bN) * K_DIM + kt + bHalf);
      const float s = scales[(n0 + bN) * (K_DIM / GROUP) + ((unsigned)(kt + bHalf) >> 7)];
      _Float16* dst = &sB[bN * LDS_PAD + bHalf];
#pragma unroll
      for (int j = 0; j < 16; ++j) dst[j] = (_Float16)(s * (float)w[j]);
    }
    __syncthreads();

    v16h aF[2];
#pragma unroll
    for (int mi = 0; mi < 2; ++mi) {
      const _Float16* p = &sA[(wm * 32 + mi * 16 + frow) * LDS_PAD + fk];
      v8h lo = *reinterpret_cast<const v8h*>(p);
      v8h hi = *reinterpret_cast<const v8h*>(p + 16);
      aF[mi] = __builtin_shufflevector(lo, hi, 0,1,2,3,4,5,6,7,8,9,10,11,12,13,14,15);
    }
    v16h bF[4];
#pragma unroll
    for (int ni = 0; ni < 4; ++ni) {
      const _Float16* p = &sB[(wn * 64 + ni * 16 + frow) * LDS_PAD + fk];
      v8h lo = *reinterpret_cast<const v8h*>(p);
      v8h hi = *reinterpret_cast<const v8h*>(p + 16);
      bF[ni] = __builtin_shufflevector(lo, hi, 0,1,2,3,4,5,6,7,8,9,10,11,12,13,14,15);
    }
#pragma unroll
    for (int mi = 0; mi < 2; ++mi)
#pragma unroll
      for (int ni = 0; ni < 4; ++ni)
        acc[mi][ni] = __builtin_amdgcn_wmma_f32_16x16x32_f16(
            false, aF[mi], false, bF[ni], (short)0, acc[mi][ni], false, false);
    __syncthreads();
  }

#pragma unroll
  for (int mi = 0; mi < 2; ++mi) {
#pragma unroll
    for (int ni = 0; ni < 4; ++ni) {
      const size_t row_b = m0 + wm * 32 + mi * 16 + ((lane >> 4) << 3);
      const size_t col   = n0 + wn * 64 + ni * 16 + (lane & 15);
#pragma unroll
      for (int r = 0; r < 8; ++r)
        out[(row_b + r) * N_DIM + col] = acc[mi][ni][r];
    }
  }
}

extern "C" void kernel_launch(void* const* d_in, const int* in_sizes, int n_in,
                              void* d_out, int out_size, void* d_ws, size_t ws_size,
                              hipStream_t stream) {
  (void)in_sizes; (void)n_in; (void)out_size;
  const float*  xin    = (const float*)d_in[0];
  const int8_t* tern   = (const int8_t*)d_in[1];
  const float*  scales = (const float*)d_in[2];
  float*        out    = (float*)d_out;

  const size_t needA = (size_t)M_DIM * K_DIM * sizeof(_Float16);  // 64 MiB
  const size_t needB = (size_t)N_DIM * K_DIM * sizeof(_Float16);  // 32 MiB

  if (ws_size >= needA + needB) {
    _Float16* xw = (_Float16*)d_ws;
    _Float16* ww = xw + (size_t)M_DIM * K_DIM;
    cvt_x_f16<<<(M_DIM * (size_t)K_DIM) / (256 * 8), 256, 0, stream>>>(xin, xw);
    dequant_w_f16<<<((size_t)N_DIM * K_DIM) / (256 * 16), 256, 0, stream>>>(tern, scales, ww);
    dim3 grid(N_DIM / BN, M_DIM / BM);   // 32 x 32 = 1024 blocks
    gemm_f16_async<<<grid, 256, 0, stream>>>(xw, ww, out);
  } else {
    dim3 grid(N_DIM / 128, M_DIM / 128); // 32 x 64 = 2048 blocks
    ternary_linear_wmma<<<grid, 256, 0, stream>>>(xin, tern, scales, out);
  }
}
